// SelfAttentionHead_32195074851091
// MI455X (gfx1250) — matrix operand order repaired and checked by reading, not compiled, not run on hardware
//
#include <hip/hip_runtime.h>
#include <hip/hip_bf16.h>

typedef _Float16 v16h __attribute__((ext_vector_type(16)));
typedef float    v8f  __attribute__((ext_vector_type(8)));

#define B_   8
#define T_   2048
#define C_   1024
#define H_   64
#define BTH_ (B_ * T_ * H_)

// ---- layout helpers -------------------------------------------------------
// B-matrix (32x16 f16) per-lane data: two contiguous 8-element K chunks of one
// column: elements [0..7] at p[0..7], elements [8..15] at p[16..23].
__device__ __forceinline__ v16h load_b16_f16(const float* __restrict__ p) {
    float4 f0 = *(const float4*)(p);
    float4 f1 = *(const float4*)(p + 4);
    float4 f2 = *(const float4*)(p + 16);
    float4 f3 = *(const float4*)(p + 20);
    v16h r;
    r[0]=(_Float16)f0.x; r[1]=(_Float16)f0.y; r[2]=(_Float16)f0.z; r[3]=(_Float16)f0.w;
    r[4]=(_Float16)f1.x; r[5]=(_Float16)f1.y; r[6]=(_Float16)f1.z; r[7]=(_Float16)f1.w;
    r[8]=(_Float16)f2.x; r[9]=(_Float16)f2.y; r[10]=(_Float16)f2.z; r[11]=(_Float16)f2.w;
    r[12]=(_Float16)f3.x; r[13]=(_Float16)f3.y; r[14]=(_Float16)f3.z; r[15]=(_Float16)f3.w;
    return r;
}

// A-matrix (16x32 f16) per-lane data: two contiguous 8-element K chunks of one
// row: elements [0..7] at p[0..7], elements [8..15] at p[16..23].
__device__ __forceinline__ v16h load_a16_f16(const float* __restrict__ p, float scale) {
    float4 f0 = *(const float4*)(p);
    float4 f1 = *(const float4*)(p + 4);
    float4 f2 = *(const float4*)(p + 16);
    float4 f3 = *(const float4*)(p + 20);
    v16h r;
    r[0]=(_Float16)(f0.x*scale); r[1]=(_Float16)(f0.y*scale);
    r[2]=(_Float16)(f0.z*scale); r[3]=(_Float16)(f0.w*scale);
    r[4]=(_Float16)(f1.x*scale); r[5]=(_Float16)(f1.y*scale);
    r[6]=(_Float16)(f1.z*scale); r[7]=(_Float16)(f1.w*scale);
    r[8]=(_Float16)(f2.x*scale); r[9]=(_Float16)(f2.y*scale);
    r[10]=(_Float16)(f2.z*scale); r[11]=(_Float16)(f2.w*scale);
    r[12]=(_Float16)(f3.x*scale); r[13]=(_Float16)(f3.y*scale);
    r[14]=(_Float16)(f3.z*scale); r[15]=(_Float16)(f3.w*scale);
    return r;
}

// ---- kernel 1: KQV projections -------------------------------------------
// grid = (12, B*T/16); one wave per 16x16 output tile of [B*T, 192].
// Column tiles are the fastest-varying block index so the 12 re-reads of each
// x tile hit WGP$/L2. K,Q stored [B,T,H]; V stored transposed [B,H,T] so the
// attention kernel's P@V B-tiles are contiguous loads.
__global__ __launch_bounds__(32)
void proj_kernel(const float* __restrict__ x,
                 const float* __restrict__ Wk,
                 const float* __restrict__ Wq,
                 const float* __restrict__ Wv,
                 float* __restrict__ ws)
{
    const int lane = threadIdx.x & 31;
    const int mrow = lane & 15;
    const int hf   = lane >> 4;

    const int n0   = blockIdx.x * 16;          // 0..176
    const int bt0  = blockIdx.y * 16;
    const int mat  = n0 >> 6;                  // 0:K 1:Q 2:V
    const int col0 = n0 & 63;
    const float* W = (mat == 0) ? Wk : (mat == 1) ? Wq : Wv;

    const float* xrow = x + (size_t)(bt0 + mrow) * C_;       // A row m
    const float* wrow = W + (size_t)(col0 + mrow) * C_;      // B column n

    v8f acc = {};
    for (int k0 = 0; k0 < C_; k0 += 32) {
        v16h a = load_a16_f16(xrow + k0 + (hf << 3), 1.0f);
        v16h b = load_b16_f16(wrow + k0 + (hf << 3));
        acc = __builtin_amdgcn_wmma_f32_16x16x32_f16(
                  false, a, false, b, (short)0, acc, false, false);
    }

    if (mat == 2) {
        // V transposed: Vt[b][h][t]; rows r are consecutive t -> vector stores
        const int b = bt0 / T_;
        const int t = (bt0 % T_) + (hf << 3);
        const int h = col0 + mrow;
        float* vt = ws + (size_t)2 * BTH_ + ((size_t)b * H_ + h) * T_ + t;
        float4 lo = { acc[0], acc[1], acc[2], acc[3] };
        float4 hi = { acc[4], acc[5], acc[6], acc[7] };
        *(float4*)(vt)     = lo;
        *(float4*)(vt + 4) = hi;
    } else {
        float* outp = ws + (size_t)mat * BTH_;
#pragma unroll
        for (int r = 0; r < 8; ++r) {
            const int row = bt0 + r + (hf << 3);
            outp[(size_t)row * H_ + col0 + mrow] = acc[r];
        }
    }
}

// ---- kernel 2: streaming causal attention (flash-style) ------------------
// grid = (T/16, B); one wave handles 16 queries, streams keys in blocks of 32.
__global__ __launch_bounds__(32)
void attn_kernel(const float* __restrict__ ws, float* __restrict__ out)
{
    __shared__ _Float16 sP[16 * 32];   // P tile, C-layout -> A-layout bounce

    const float* Kp = ws;
    const float* Qp = ws + (size_t)BTH_;
    const float* Vt = ws + (size_t)2 * BTH_;   // [B, H, T]

    const int lane = threadIdx.x & 31;
    const int mrow = lane & 15;
    const int hf   = lane >> 4;
    const int b    = blockIdx.y;
    const int t0   = blockIdx.x * 16;

    const float NEG_INF = -__builtin_inff();
    const float qscale  = 0.03125f;    // 1/sqrt(C) = 1/32

    // Q tile in A-layout, two 32-wide h-steps, pre-scaled.
    const float* qrow = Qp + ((size_t)b * T_ + t0 + mrow) * H_;
    const v16h qa0 = load_a16_f16(qrow + 0  + (hf << 3), qscale);
    const v16h qa1 = load_a16_f16(qrow + 32 + (hf << 3), qscale);

    const float* vtb = Vt + (size_t)b * H_ * T_;

    float mi[8], li[8], sc[8];
#pragma unroll
    for (int r = 0; r < 8; ++r) { mi[r] = NEG_INF; li[r] = 0.0f; }
    v8f o[4] = {};

    const int nblk = ((t0 + 15) >> 5) + 1;
    for (int blk = 0; blk < nblk; ++blk) {
        const int kb = blk << 5;

        // ---- scores: two 16-key tiles, contraction over H in 2 steps
        const float* krow0 = Kp + ((size_t)b * T_ + kb + mrow) * H_;
        const float* krow1 = krow0 + 16 * H_;
        v8f s0 = {}, s1 = {};
        s0 = __builtin_amdgcn_wmma_f32_16x16x32_f16(false, qa0, false,
                 load_b16_f16(krow0 + 0  + (hf << 3)), (short)0, s0, false, false);
        s0 = __builtin_amdgcn_wmma_f32_16x16x32_f16(false, qa1, false,
                 load_b16_f16(krow0 + 32 + (hf << 3)), (short)0, s0, false, false);
        s1 = __builtin_amdgcn_wmma_f32_16x16x32_f16(false, qa0, false,
                 load_b16_f16(krow1 + 0  + (hf << 3)), (short)0, s1, false, false);
        s1 = __builtin_amdgcn_wmma_f32_16x16x32_f16(false, qa1, false,
                 load_b16_f16(krow1 + 32 + (hf << 3)), (short)0, s1, false, false);

        // prefetch next key block (global_prefetch_b8); uniform branch
        if (blk + 1 < nblk) {
            __builtin_prefetch(krow0 + 32 * H_, 0, 3);
            __builtin_prefetch(krow1 + 32 * H_, 0, 3);
            __builtin_prefetch(vtb + (size_t)(lane * 2) * T_ + kb + 32, 0, 3);
        }

        // ---- causal mask + online softmax (row = t0 + r + hf*8, col = kb(+16)+mrow)
#pragma unroll
        for (int r = 0; r < 8; ++r) {
            const int gr = t0 + r + (hf << 3);
            float v0 = (kb + mrow      > gr) ? NEG_INF : s0[r];
            float v1 = (kb + 16 + mrow > gr) ? NEG_INF : s1[r];
            float mx = fmaxf(v0, v1);
            mx = fmaxf(mx, __shfl_xor(mx, 1, 32));
            mx = fmaxf(mx, __shfl_xor(mx, 2, 32));
            mx = fmaxf(mx, __shfl_xor(mx, 4, 32));
            mx = fmaxf(mx, __shfl_xor(mx, 8, 32));
            const float nm = fmaxf(mi[r], mx);
            const float s  = __expf(mi[r] - nm);
            const float p0 = __expf(v0 - nm);
            const float p1 = __expf(v1 - nm);
            float rs = p0 + p1;
            rs += __shfl_xor(rs, 1, 32);
            rs += __shfl_xor(rs, 2, 32);
            rs += __shfl_xor(rs, 4, 32);
            rs += __shfl_xor(rs, 8, 32);
            li[r] = li[r] * s + rs;
            mi[r] = nm;
            sc[r] = s;
            const int rl = r + (hf << 3);
            sP[rl * 32 + mrow]      = (_Float16)p0;
            sP[rl * 32 + 16 + mrow] = (_Float16)p1;
        }
        __syncthreads();

        // ---- P (A-layout) from LDS
        v16h pa;
#pragma unroll
        for (int j = 0; j < 16; ++j)
            pa[j] = sP[mrow * 32 + ((j >> 3) << 4) + (hf << 3) + (j & 7)];

        // ---- rescale accumulators
#pragma unroll
        for (int r = 0; r < 8; ++r) {
            o[0][r] *= sc[r]; o[1][r] *= sc[r];
            o[2][r] *= sc[r]; o[3][r] *= sc[r];
        }

        // ---- O += P @ V  (V^T rows are contiguous keys -> B-layout direct)
#pragma unroll
        for (int ht = 0; ht < 4; ++ht) {
            v16h vb = load_b16_f16(vtb + (size_t)(ht * 16 + mrow) * T_
                                       + kb + (hf << 3));
            o[ht] = __builtin_amdgcn_wmma_f32_16x16x32_f16(
                        false, pa, false, vb, (short)0, o[ht], false, false);
        }
        __syncthreads();
    }

    // ---- normalize and store
#pragma unroll
    for (int r = 0; r < 8; ++r) {
        const float inv = 1.0f / li[r];
        const int row = t0 + r + (hf << 3);
        const size_t base = ((size_t)b * T_ + row) * H_;
#pragma unroll
        for (int ht = 0; ht < 4; ++ht)
            out[base + ht * 16 + mrow] = o[ht][r] * inv;
    }
}

extern "C" void kernel_launch(void* const* d_in, const int* in_sizes, int n_in,
                              void* d_out, int out_size, void* d_ws, size_t ws_size,
                              hipStream_t stream) {
    const float* x  = (const float*)d_in[0];
    const float* Wk = (const float*)d_in[1];
    const float* Wq = (const float*)d_in[2];
    const float* Wv = (const float*)d_in[3];
    float* ws  = (float*)d_ws;          // K | Q | V^T staging: 3 * 4 MB = 12 MB
    float* out = (float*)d_out;

    proj_kernel<<<dim3(12, B_ * T_ / 16, 1), dim3(32, 1, 1), 0, stream>>>(
        x, Wk, Wq, Wv, ws);
    attn_kernel<<<dim3(T_ / 16, B_, 1), dim3(32, 1, 1), 0, stream>>>(ws, out);
}
